// I2P_CrossAttention_21139829030940
// MI455X (gfx1250) — matrix operand order repaired and checked by reading, not compiled, not run on hardware
//
#include <hip/hip_runtime.h>

// ---------------------------------------------------------------------------
// I2P cross-attention for MI455X (gfx1250, wave32, WMMA bf16 16x16x32)
// B=8, C=512, N=4096, M=W*H=4096
// ---------------------------------------------------------------------------

typedef __attribute__((ext_vector_type(16))) __bf16 v16bf;
typedef __attribute__((ext_vector_type(8)))  float  v8f;
typedef unsigned short bfraw;   // bf16 bits for host-side pointer plumbing

#define B_DIM 8
#define C_DIM 512
#define N_DIM 4096
#define M_DIM 4096

static __device__ __forceinline__ __bf16 f2bf(float f) { return (__bf16)f; }

union Frag16 { v16bf v; __bf16 e[16]; uint4 q[2]; };

// per-lane fragment load: two b128 at p[0..7] and p[16..23]
static __device__ __forceinline__ void ldF(const __bf16* p, Frag16& f) {
  const uint4* q = reinterpret_cast<const uint4*>(p);
  f.q[0] = q[0];
  f.q[1] = q[2];
}

// D = A(16x32 bf16) * B(32x16 bf16) + C(16x16 f32)
static __device__ __forceinline__ v8f wmma_bf16(v16bf a, v16bf b, v8f c) {
  return __builtin_amdgcn_wmma_f32_16x16x32_bf16(
      /*neg_a=*/false, a, /*neg_b=*/false, b,
      /*c_mod=*/(short)0, c, /*reuse_a=*/false, /*reuse_b=*/false);
}

// ---------------------------------------------------------------------------
// Kernel 0: f32 -> bf16 weight conversion (W is tiny, L2-resident afterwards)
// ---------------------------------------------------------------------------
__global__ void wcvt_kernel(const float* __restrict__ src,
                            bfraw* __restrict__ dst, int n) {
  int i = blockIdx.x * blockDim.x + threadIdx.x;
  if (i < n) {
    union { __bf16 h; bfraw u; } cv;
    cv.h = f2bf(src[i]);
    dst[i] = cv.u;
  }
}

// ---------------------------------------------------------------------------
// Kernel 1: projection  Y[b,n,co] = sum_c X[b,c,n] * W[co,c] + bias[co]
//   X: f32 [B, C, NN]  (A accessed "transposed": lane->n is contiguous)
//   W: bf16 [C, C] row-major
//   TRANS==0: Y bf16 [B, NN, C]    (q, k)
//   TRANS==1: Y bf16 [B, C, NN]    (v stored transposed for attention)
// One wave computes a 16-row x 512-col block: A frags register-resident;
// all 16 B-fragments of a tile are loaded up-front for deep MLP.
// ---------------------------------------------------------------------------
template <int TRANS>
__global__ __launch_bounds__(256) void proj_kernel(
    const float* __restrict__ X, const bfraw* __restrict__ Wraw,
    const float* __restrict__ bias, bfraw* __restrict__ Yraw, int NN) {
  const __bf16* Wb = reinterpret_cast<const __bf16*>(Wraw);
  __bf16* Y = reinterpret_cast<__bf16*>(Yraw);

  const int lane = threadIdx.x & 31;
  const int wid  = (blockIdx.x * blockDim.x + threadIdx.x) >> 5;
  const int tilesPerB = NN >> 4;
  const int b  = wid / tilesPerB;
  const int n0 = (wid % tilesPerB) << 4;
  const int half = lane >> 4;
  const int l16  = lane & 15;
  const int nl   = n0 + l16;

  // A fragments: 16 K-slices of 32 along C. A layout (16-bit, 16x32):
  // lane<16: K = {0..7, 16..23}; lane>=16: K = {8..15, 24..31} (per slice).
  Frag16 af[16];
#pragma unroll
  for (int s = 0; s < 16; ++s) {
    const int kk = s << 5;
#pragma unroll
    for (int e = 0; e < 8; ++e) {
      const int c0 = kk + half * 8 + e;
      const int c1 = kk + 16 + half * 8 + e;
      af[s].e[e]     = f2bf(X[((size_t)b * C_DIM + c0) * NN + nl]);
      af[s].e[e + 8] = f2bf(X[((size_t)b * C_DIM + c1) * NN + nl]);
    }
  }

  for (int t = 0; t < 32; ++t) {   // 32 output-column tiles of 16
    const int co = (t << 4) + l16;
    const __bf16* wrow = &Wb[(size_t)co * C_DIM + half * 8];

    // issue all 32 b128 loads first (deep MLP), then drain into WMMA chain
    Frag16 wb[16];
#pragma unroll
    for (int s = 0; s < 16; ++s) ldF(wrow + (s << 5), wb[s]);

    v8f acc = {};
#pragma unroll
    for (int s = 0; s < 16; ++s) acc = wmma_bf16(af[s].v, wb[s].v, acc);

    const float bv = bias[co];
    if (TRANS == 0) {
#pragma unroll
      for (int j = 0; j < 8; ++j) {
        const int n = n0 + j + 8 * half;   // C layout: row = j + 8*(lane/16)
        Y[((size_t)b * NN + n) * C_DIM + co] = f2bf(acc[j] + bv);
      }
    } else {
      // fixed co per lane, n consecutive over j -> pack 2 bf16 per store
#pragma unroll
      for (int j = 0; j < 8; j += 2) {
        const int n = n0 + 8 * half + j;
        union { __bf16 h[2]; unsigned u; } pk;
        pk.h[0] = f2bf(acc[j] + bv);
        pk.h[1] = f2bf(acc[j + 1] + bv);
        *reinterpret_cast<unsigned*>(&Y[((size_t)b * C_DIM + co) * NN + n]) =
            pk.u;
      }
    }
  }
}

// ---------------------------------------------------------------------------
// Kernel 2: flash attention.
// One wave = 16 queries, full C=512. Q frags resident (128 VGPRs),
// ctx accumulators 32 x v8f (256 VGPRs).
// Pipelining:
//   * K fragments: depth-4 rotation inside the S phase, PLUS cross-iteration
//     preload of the next tile's first 4 slices issued right after the P
//     store (hidden under rescale + the 32-WMMA V phase).
//   * V fragments: depth-8 rotation (16 loads in flight), preloads hoisted
//     above the softmax so its VALU work hides their latency.
//   * P LDS stores issued before the acc rescale; s_wait_dscnt 0 after
//     (DS is wave-in-order on CDNA5).
// scale = C^-0.5 folded into final 1/l normalization.
// ---------------------------------------------------------------------------
__global__ __launch_bounds__(256) void attn_kernel(
    const bfraw* __restrict__ Qraw,   // [B, N, C]
    const bfraw* __restrict__ Kraw,   // [B, M, C]
    const bfraw* __restrict__ VTraw,  // [B, C, M]  (transposed V)
    bfraw* __restrict__ CTXraw) {     // [B, N, C]
  const __bf16* Q  = reinterpret_cast<const __bf16*>(Qraw);
  const __bf16* K  = reinterpret_cast<const __bf16*>(Kraw);
  const __bf16* VT = reinterpret_cast<const __bf16*>(VTraw);
  __bf16* CTX = reinterpret_cast<__bf16*>(CTXraw);

  __shared__ __bf16 ldsP[8 * 16 * 32];   // 8 waves x (16 rows x 32 m) bf16

  const int lane  = threadIdx.x & 31;
  const int wslot = threadIdx.x >> 5;
  const int wid   = (blockIdx.x * blockDim.x + threadIdx.x) >> 5;
  const int b  = wid >> 8;               // N/16 = 256 query tiles per batch
  const int n0 = (wid & 255) << 4;
  const int half = lane >> 4;
  const int l16  = lane & 15;

  // Resident Q fragments (16 K-slices)
  Frag16 qf[16];
  {
    const __bf16* qrow = &Q[((size_t)b * N_DIM + n0 + l16) * C_DIM];
#pragma unroll
    for (int s = 0; s < 16; ++s) {
      const int kk = s << 5;
      qf[s].q[0] = *reinterpret_cast<const uint4*>(qrow + kk + half * 8);
      qf[s].q[1] = *reinterpret_cast<const uint4*>(qrow + kk + 16 + half * 8);
    }
  }

  v8f acc[32];
#pragma unroll
  for (int t = 0; t < 32; ++t) acc[t] = (v8f){};

  float mrow[8], lrow[8];
#pragma unroll
  for (int j = 0; j < 8; ++j) { mrow[j] = -3.0e38f; lrow[j] = 0.0f; }

  __bf16* pbase = &ldsP[wslot * 512];
  const __bf16* vbat  = &VT[((size_t)b * C_DIM + l16) * M_DIM + half * 8];
  const __bf16* kbat  = &K[((size_t)b * M_DIM + l16) * C_DIM + half * 8];

  // Cross-iteration K pipeline: prologue loads for m0 = 0
  Frag16 kb0[4], kb1[4];
#pragma unroll
  for (int s = 0; s < 4; ++s) {
    ldF(kbat + (s << 5), kb0[s]);
    ldF(kbat + (size_t)16 * C_DIM + (s << 5), kb1[s]);
  }

  for (int m0 = 0; m0 < M_DIM; m0 += 32) {
    const __bf16* krow0 = kbat + (size_t)m0 * C_DIM;
    const __bf16* krow1 = krow0 + (size_t)16 * C_DIM;

    // ---- S = Q * K^T  (two 16x16 tiles over the 32-key block)
    v8f s0 = {}, s1 = {};
#pragma unroll
    for (int s = 0; s < 16; ++s) {
      s0 = wmma_bf16(qf[s].v, kb0[s & 3].v, s0);
      s1 = wmma_bf16(qf[s].v, kb1[s & 3].v, s1);
      if (s + 4 < 16) {
        ldF(krow0 + ((s + 4) << 5), kb0[s & 3]);
        ldF(krow1 + ((s + 4) << 5), kb1[s & 3]);
      }
    }

    // ---- preload V fragments (depth 8); softmax below hides their latency
    Frag16 vb[8];
#pragma unroll
    for (int t = 0; t < 8; ++t)
      ldF(vbat + ((size_t)(t << 4)) * M_DIM + m0, vb[t]);

    // ---- online softmax (row r = j + 8*half lives across a 16-lane half)
    float p0[8], p1[8], alpha[8];
    bool anyresc = false;
#pragma unroll
    for (int j = 0; j < 8; ++j) {
      float t = fmaxf(s0[j], s1[j]);
      t = fmaxf(t, __shfl_xor(t, 1, 32));
      t = fmaxf(t, __shfl_xor(t, 2, 32));
      t = fmaxf(t, __shfl_xor(t, 4, 32));
      t = fmaxf(t, __shfl_xor(t, 8, 32));
      const float mnew = fmaxf(mrow[j], t);
      alpha[j] = (mnew == mrow[j]) ? 1.0f : __expf(mrow[j] - mnew);
      anyresc |= (alpha[j] != 1.0f);
      p0[j] = __expf(s0[j] - mnew);
      p1[j] = __expf(s1[j] - mnew);
      float rs = p0[j] + p1[j];
      rs += __shfl_xor(rs, 1, 32);
      rs += __shfl_xor(rs, 2, 32);
      rs += __shfl_xor(rs, 4, 32);
      rs += __shfl_xor(rs, 8, 32);
      lrow[j] = lrow[j] * alpha[j] + rs;
      mrow[j] = mnew;
    }

    // ---- P stores first: DS latency hides under K-preload + rescale below
#pragma unroll
    for (int j = 0; j < 8; ++j) {
      const int r = j + 8 * half;
      pbase[r * 32 + l16]      = f2bf(p0[j]);
      pbase[r * 32 + 16 + l16] = f2bf(p1[j]);
    }

    // ---- cross-iteration: issue next tile's first K fragments now; they
    // have the whole rescale + 32-WMMA V phase to complete.
    if (m0 + 32 < M_DIM) {
      const __bf16* nk0 = kbat + (size_t)(m0 + 32) * C_DIM;
      const __bf16* nk1 = nk0 + (size_t)16 * C_DIM;
#pragma unroll
      for (int s = 0; s < 4; ++s) {
        ldF(nk0 + (s << 5), kb0[s]);
        ldF(nk1 + (s << 5), kb1[s]);
      }
    }

    if (__ballot(anyresc) != 0ull) {
#pragma unroll
      for (int t = 0; t < 32; ++t) {
#pragma unroll
        for (int j = 0; j < 8; ++j) acc[t][j] = acc[t][j] * alpha[j];
      }
    }

    asm volatile("s_wait_dscnt 0" ::: "memory");  // DS wave-in-order + wait

    Frag16 pf;
    pf.q[0] = *reinterpret_cast<const uint4*>(pbase + l16 * 32 + half * 8);
    pf.q[1] = *reinterpret_cast<const uint4*>(pbase + l16 * 32 + 16 + half * 8);

    // ---- ctx += P(16x32) * V(32x16) for all 32 C-tiles (depth-8 pipeline)
#pragma unroll
    for (int t = 0; t < 32; ++t) {
      acc[t] = wmma_bf16(pf.v, vb[t & 7].v, acc[t]);
      if (t + 8 < 32)
        ldF(vbat + ((size_t)((t + 8) << 4)) * M_DIM + m0, vb[t & 7]);
    }
  }

  // ---- finalize: ctx * (C^-0.5 / l)   (softmax then scale, per reference)
  const float scale = 0.04419417382415922f;  // 1/sqrt(512)
#pragma unroll
  for (int j = 0; j < 8; ++j) lrow[j] = scale / lrow[j];
#pragma unroll
  for (int t = 0; t < 32; ++t) {
#pragma unroll
    for (int j = 0; j < 8; ++j) {
      const int n = n0 + j + 8 * half;
      CTX[((size_t)b * N_DIM + n) * C_DIM + (t << 4) + l16] =
          f2bf(acc[t][j] * lrow[j]);
    }
  }
}

// ---------------------------------------------------------------------------
// Kernel 3: out[b,co,n] = sum_c ctx[b,n,c]*Wo[co,c] + bo[co] + feat0[b,co,n]
// A = ctx bf16 (contiguous per-lane frag loads), f32 transposed store with
// float4 vectors (each 64B line covered by lane pair l / l+16).
// ---------------------------------------------------------------------------
__global__ __launch_bounds__(256) void oproj_kernel(
    const bfraw* __restrict__ CTXraw, const bfraw* __restrict__ Woraw,
    const float* __restrict__ bo, const float* __restrict__ feat0,
    float* __restrict__ out) {
  const __bf16* CTX = reinterpret_cast<const __bf16*>(CTXraw);
  const __bf16* Wb  = reinterpret_cast<const __bf16*>(Woraw);

  const int lane = threadIdx.x & 31;
  const int wid  = (blockIdx.x * blockDim.x + threadIdx.x) >> 5;
  const int b  = wid >> 8;
  const int n0 = (wid & 255) << 4;
  const int half = lane >> 4;
  const int l16  = lane & 15;

  Frag16 af[16];
  {
    const __bf16* arow = &CTX[((size_t)b * N_DIM + n0 + l16) * C_DIM];
#pragma unroll
    for (int s = 0; s < 16; ++s) {
      const int kk = s << 5;
      af[s].q[0] = *reinterpret_cast<const uint4*>(arow + kk + half * 8);
      af[s].q[1] = *reinterpret_cast<const uint4*>(arow + kk + 16 + half * 8);
    }
  }

  for (int t = 0; t < 32; ++t) {
    const int co = (t << 4) + l16;
    const __bf16* wrow = &Wb[(size_t)co * C_DIM + half * 8];

    Frag16 wb[16];
#pragma unroll
    for (int s = 0; s < 16; ++s) ldF(wrow + (s << 5), wb[s]);

    v8f acc = {};
#pragma unroll
    for (int s = 0; s < 16; ++s) acc = wmma_bf16(af[s].v, wb[s].v, acc);

    const float bv = bo[co];
    const size_t base = ((size_t)b * C_DIM + co) * N_DIM + n0 + 8 * half;
    const float4 r0 = *reinterpret_cast<const float4*>(&feat0[base]);
    const float4 r1 = *reinterpret_cast<const float4*>(&feat0[base + 4]);
    float4 o0, o1;
    o0.x = acc[0] + bv + r0.x; o0.y = acc[1] + bv + r0.y;
    o0.z = acc[2] + bv + r0.z; o0.w = acc[3] + bv + r0.w;
    o1.x = acc[4] + bv + r1.x; o1.y = acc[5] + bv + r1.y;
    o1.z = acc[6] + bv + r1.z; o1.w = acc[7] + bv + r1.w;
    *reinterpret_cast<float4*>(&out[base])     = o0;
    *reinterpret_cast<float4*>(&out[base + 4]) = o1;
  }
}

// ---------------------------------------------------------------------------
// Host launcher
// ---------------------------------------------------------------------------
extern "C" void kernel_launch(void* const* d_in, const int* in_sizes, int n_in,
                              void* d_out, int out_size, void* d_ws,
                              size_t ws_size, hipStream_t stream) {
  const float* feat0 = (const float*)d_in[0];
  const float* feat1 = (const float*)d_in[1];
  const float* Wq = (const float*)d_in[2];
  const float* bq = (const float*)d_in[3];
  const float* Wk = (const float*)d_in[4];
  const float* bk = (const float*)d_in[5];
  const float* Wv = (const float*)d_in[6];
  const float* bv = (const float*)d_in[7];
  const float* Wo = (const float*)d_in[8];
  const float* bo = (const float*)d_in[9];
  float* out = (float*)d_out;

  char* ws = (char*)d_ws;
  const size_t WSZ = (size_t)C_DIM * C_DIM * sizeof(bfraw);          // 0.5 MB
  const size_t QSZ = (size_t)B_DIM * N_DIM * C_DIM * sizeof(bfraw);  // 32 MB
  bfraw* Wqb  = (bfraw*)(ws + 0 * WSZ);
  bfraw* Wkb  = (bfraw*)(ws + 1 * WSZ);
  bfraw* Wvb  = (bfraw*)(ws + 2 * WSZ);
  bfraw* Wob  = (bfraw*)(ws + 3 * WSZ);
  bfraw* qb   = (bfraw*)(ws + 4 * WSZ);
  bfraw* kb   = (bfraw*)(ws + 4 * WSZ + 1 * QSZ);
  bfraw* vTb  = (bfraw*)(ws + 4 * WSZ + 2 * QSZ);
  bfraw* ctxb = (bfraw*)(ws + 4 * WSZ + 3 * QSZ);

  const int WN = C_DIM * C_DIM;  // 262144
  wcvt_kernel<<<WN / 256, 256, 0, stream>>>(Wq, Wqb, WN);
  wcvt_kernel<<<WN / 256, 256, 0, stream>>>(Wk, Wkb, WN);
  wcvt_kernel<<<WN / 256, 256, 0, stream>>>(Wv, Wvb, WN);
  wcvt_kernel<<<WN / 256, 256, 0, stream>>>(Wo, Wob, WN);

  // 2048 waves per projection: 256 blocks x 8 waves
  proj_kernel<0><<<256, 256, 0, stream>>>(feat0, Wqb, bq, qb,  N_DIM);
  proj_kernel<0><<<256, 256, 0, stream>>>(feat1, Wkb, bk, kb,  M_DIM);
  proj_kernel<1><<<256, 256, 0, stream>>>(feat1, Wvb, bv, vTb, M_DIM);

  attn_kernel<<<256, 256, 0, stream>>>(qb, kb, vTb, ctxb);

  oproj_kernel<<<256, 256, 0, stream>>>(ctxb, Wob, bo, feat0, out);
}
